// ResDeepGCN_32770600468621
// MI455X (gfx1250) — matrix operand order repaired and checked by reading, not compiled, not run on hardware
//
#include <hip/hip_runtime.h>

typedef __attribute__((ext_vector_type(16))) _Float16 v16h;
typedef __attribute__((ext_vector_type(8)))  _Float16 v8h;
typedef __attribute__((ext_vector_type(8)))  float    v8f;

#define BM 64
#define BN 128
#define BK 32
#define LDT 40   // LDS row stride in halves (80 B: 16B-aligned chunks, conflict-free frag reads)

// Generic GEMM: C[b] = epilogue( A[b] (MxK,f32) * B[b] (KxN,f32) )
// flags bit0: add bias[n] then ReLU; bit1: add residual R[b][m][n] (after ReLU).
// f32 -> f16 conversion during LDS staging; f32 accumulate via v_wmma_f32_16x16x32_f16.
// 8 waves/block, each wave owns a 32x32 tile (2x2 fragments, 4 WMMA per K-step).
// Double-buffered LDS: global loads for tile kt+1 overlap WMMAs on tile kt.
__global__ __launch_bounds__(256) void gemm_wmma_f16(
    const float* __restrict__ A, long sAb, int lda,
    const float* __restrict__ Bm, long sBb, int ldb,
    const float* __restrict__ bias,
    const float* __restrict__ R, long sRb, int ldr,
    float* __restrict__ C, long sCb, int ldc,
    int M, int N, int K, int flags)
{
    __shared__ _Float16 As[2][BM * LDT];   // [m][k]  2 x 5 KB
    __shared__ _Float16 Bs[2][BN * LDT];   // [n][k]  2 x 10 KB (B stored transposed)

    const int tid  = threadIdx.x;
    const int lane = tid & 31;
    const int wave = tid >> 5;
    const int wm   = (wave & 1) * 32;   // wave tile row within block
    const int wn   = (wave >> 1) * 32;  // wave tile col within block
    const int m0   = blockIdx.y * BM;
    const int n0   = blockIdx.x * BN;
    const int bz   = blockIdx.z;

    A  += (long)bz * sAb;
    Bm += (long)bz * sBb;
    C  += (long)bz * sCb;
    if (R) R += (long)bz * sRb;

    const int lr    = lane & 15;
    const int hi16  = lane >> 4;
    const int akoff = hi16 * 8;    // A frag: K chunks {0..7,16..23} / {8..15,24..31}
    const int bkoff = hi16 * 8;    // B frag: K chunks {0..7,16..23} / {8..15,24..31}

    // ---- loop-invariant staging coordinates ----
    int ar[2], ac[2];  const float* aBase[2];  bool aOk[2];
    #pragma unroll
    for (int i = 0; i < 2; ++i) {
        int idx = tid + i * 256;            // 512 float4 cover BM x BK
        ar[i] = idx >> 3;                   // row 0..63
        ac[i] = (idx & 7) << 2;             // col 0,4,..28
        aOk[i]   = (m0 + ar[i]) < M;
        aBase[i] = A + (long)(m0 + ar[i]) * lda + ac[i];
    }
    int br[4], bc[4];  const float* bBase[4];  bool bOk[4];
    #pragma unroll
    for (int i = 0; i < 4; ++i) {
        int idx = tid + i * 256;            // 1024 float4 cover BK x BN
        br[i] = idx >> 5;                   // k row 0..31
        bc[i] = (idx & 31) << 2;            // n col 0,4,..124
        bOk[i]   = (n0 + bc[i]) < N;        // N % 4 == 0 -> whole float4 in/out
        bBase[i] = Bm + (long)br[i] * ldb + (n0 + bc[i]);
    }
    const bool fullM = (m0 + BM) <= M;
    const bool fullN = (n0 + BN) <= N;

    float4 ra[2], rb[4];

    // phase 1: global -> registers (fast path = no exec masking, loads back-to-back)
    auto load_tiles = [&](int k0) {
        const bool fullK = (k0 + BK) <= K;
        if (fullM & fullK) {
            #pragma unroll
            for (int i = 0; i < 2; ++i) ra[i] = *(const float4*)(aBase[i] + k0);
        } else {
            #pragma unroll
            for (int i = 0; i < 2; ++i) {
                ra[i] = make_float4(0.f, 0.f, 0.f, 0.f);
                if (aOk[i] && (k0 + ac[i]) < K)       // K % 4 == 0
                    ra[i] = *(const float4*)(aBase[i] + k0);
            }
        }
        if (fullN & fullK) {
            #pragma unroll
            for (int i = 0; i < 4; ++i) rb[i] = *(const float4*)(bBase[i] + (long)k0 * ldb);
        } else {
            #pragma unroll
            for (int i = 0; i < 4; ++i) {
                rb[i] = make_float4(0.f, 0.f, 0.f, 0.f);
                if (bOk[i] && (k0 + br[i]) < K)
                    rb[i] = *(const float4*)(bBase[i] + (long)k0 * ldb);
            }
        }
    };

    // phase 2: convert f32->f16 and store into LDS buffer `buf`
    auto store_tiles = [&](int buf) {
        #pragma unroll
        for (int i = 0; i < 2; ++i) {
            _Float16* p = &As[buf][ar[i] * LDT + ac[i]];
            p[0] = (_Float16)ra[i].x; p[1] = (_Float16)ra[i].y;
            p[2] = (_Float16)ra[i].z; p[3] = (_Float16)ra[i].w;
        }
        #pragma unroll
        for (int i = 0; i < 4; ++i) {
            Bs[buf][(bc[i] + 0) * LDT + br[i]] = (_Float16)rb[i].x;
            Bs[buf][(bc[i] + 1) * LDT + br[i]] = (_Float16)rb[i].y;
            Bs[buf][(bc[i] + 2) * LDT + br[i]] = (_Float16)rb[i].z;
            Bs[buf][(bc[i] + 3) * LDT + br[i]] = (_Float16)rb[i].w;
        }
    };

    v8f acc[2][2] = {};

    auto compute = [&](int buf) {
        v16h afrag[2], bfrag[2];
        #pragma unroll
        for (int i = 0; i < 2; ++i) {
            const _Float16* base = &As[buf][(wm + i * 16 + lr) * LDT];
            v8h lo = *(const v8h*)(base + akoff);
            v8h hi = *(const v8h*)(base + akoff + 16);
            afrag[i] = __builtin_shufflevector(lo, hi,
                0,1,2,3,4,5,6,7,8,9,10,11,12,13,14,15);
        }
        #pragma unroll
        for (int i = 0; i < 2; ++i) {
            const _Float16* base = &Bs[buf][(wn + i * 16 + lr) * LDT];
            v8h lo = *(const v8h*)(base + bkoff);
            v8h hi = *(const v8h*)(base + bkoff + 16);
            bfrag[i] = __builtin_shufflevector(lo, hi,
                0,1,2,3,4,5,6,7,8,9,10,11,12,13,14,15);
        }
        #pragma unroll
        for (int mi = 0; mi < 2; ++mi)
            #pragma unroll
            for (int ni = 0; ni < 2; ++ni)
                acc[mi][ni] = __builtin_amdgcn_wmma_f32_16x16x32_f16(
                    false, afrag[mi], false, bfrag[ni],
                    (short)0, acc[mi][ni], false, false);
    };

    // ---- software-pipelined main loop (ping-pong LDS) ----
    const int steps = (K + BK - 1) / BK;
    load_tiles(0);
    store_tiles(0);
    __syncthreads();
    for (int kt = 0; kt < steps; ++kt) {
        if (kt + 1 < steps) load_tiles((kt + 1) * BK);   // HBM loads in flight...
        compute(kt & 1);                                 // ...while WMMAs run
        if (kt + 1 < steps) store_tiles((kt + 1) & 1);   // waits land after WMMAs
        __syncthreads();
    }

    // ---- epilogue: C layout VGPR r -> M=r (+8 for lanes>=16), N=lane&15 ----
    #pragma unroll
    for (int mi = 0; mi < 2; ++mi) {
        #pragma unroll
        for (int ni = 0; ni < 2; ++ni) {
            int n = n0 + wn + ni * 16 + lr;
            if (n >= N) continue;
            int mbase = m0 + wm + mi * 16 + hi16 * 8;
            float bv = (flags & 1) ? bias[n] : 0.f;
            #pragma unroll
            for (int r = 0; r < 8; ++r) {
                int m = mbase + r;
                if (m >= M) continue;
                float v = acc[mi][ni][r] + bv;
                if (flags & 1) v = v > 0.f ? v : 0.f;
                if (flags & 2) v += R[(long)m * ldr + n];
                C[(long)m * ldc + n] = v;
            }
        }
    }
}

// out[m][0:2] = hid[m][0:64] @ Wp2[64][2] + bp2  (too skinny for WMMA)
__global__ __launch_bounds__(256) void final_linear(
    const float* __restrict__ hid, const float* __restrict__ W,
    const float* __restrict__ bias, float* __restrict__ out, int Mtot)
{
    int m = blockIdx.x * blockDim.x + threadIdx.x;
    if (m >= Mtot) return;
    float a0 = bias[0], a1 = bias[1];
    const float* h = hid + (long)m * 64;
    #pragma unroll 8
    for (int k = 0; k < 64; ++k) {
        float x = h[k];
        a0 += x * W[k * 2 + 0];
        a1 += x * W[k * 2 + 1];
    }
    out[(long)m * 2 + 0] = a0;
    out[(long)m * 2 + 1] = a1;
}

static inline void launch_gemm(hipStream_t s,
    const float* A, long sAb, int lda,
    const float* B, long sBb, int ldb,
    const float* bias,
    const float* R, long sRb, int ldr,
    float* C, long sCb, int ldc,
    int M, int N, int K, int batch, int flags)
{
    dim3 grid((N + BN - 1) / BN, (M + BM - 1) / BM, batch);
    gemm_wmma_f16<<<grid, dim3(256), 0, s>>>(
        A, sAb, lda, B, sBb, ldb, bias, R, sRb, ldr, C, sCb, ldc, M, N, K, flags);
}

extern "C" void kernel_launch(void* const* d_in, const int* in_sizes, int n_in,
                              void* d_out, int out_size, void* d_ws, size_t ws_size,
                              hipStream_t stream) {
    (void)in_sizes; (void)n_in; (void)out_size; (void)ws_size;

    const float* x      = (const float*)d_in[0];   // [64,360,360]
    const float* adj    = (const float*)d_in[1];   // [64,360,360]
    const float* W_head = (const float*)d_in[2];   // [360,256]
    const float* b_head = (const float*)d_in[3];   // [256]
    const float* W1s    = (const float*)d_in[4];   // [13,256,256]
    const float* b1s    = (const float*)d_in[5];   // [13,256]
    const float* W2s    = (const float*)d_in[6];   // [13,256,256]
    const float* b2s    = (const float*)d_in[7];   // [13,256]
    const float* Wf     = (const float*)d_in[8];   // [3584,256]
    const float* bf     = (const float*)d_in[9];   // [256]
    const float* Wp1    = (const float*)d_in[10];  // [256,64]
    const float* bp1    = (const float*)d_in[11];  // [64]
    const float* Wp2    = (const float*)d_in[12];  // [64,2]
    const float* bp2    = (const float*)d_in[13];  // [2]

    const int  NB   = 14, C1 = 256, NODES = 360, BATCH = 64;
    const long MR   = (long)BATCH * NODES;         // 23040 total rows
    const int  CF   = NB * C1;                     // 3584 fused channels

    float* ws     = (float*)d_ws;
    float* feats  = ws;                            // [MR, 3584]  (h | ys[0..12])
    float* t      = feats + MR * CF;               // [MR, 256]
    float* u      = t + MR * C1;                   // [MR, 256]
    float* hid    = u + MR * C1;                   // [MR, 64]
    float* out    = (float*)d_out;                 // [MR, 2]
    float* fusion = out + MR * 2;                  // [MR, 256]

    const long sAdj = (long)NODES * NODES;         // per-batch adj stride
    const long sT   = (long)NODES * C1;            // per-batch feature stride
    const long sF   = (long)NODES * CF;            // per-batch feats stride

    // 1) t = x @ W_head               [23040,360]x[360,256]
    launch_gemm(stream, x, 0, 360, W_head, 0, 256, nullptr,
                nullptr, 0, 0, t, 0, 256, (int)MR, 256, 360, 1, 0);
    // 2) h = relu(adj_b @ t_b + b_head) -> feats col-block 0 (batched over 64)
    launch_gemm(stream, adj, sAdj, 360, t, sT, 256, b_head,
                nullptr, 0, 0, feats, sF, CF, NODES, 256, 360, BATCH, 1);

    for (int k = 0; k < NB - 1; ++k) {
        const float* hk = feats + (long)k * C1;     // carry = feats col-block k
        // 3a) t = h_k @ W1s[k]
        launch_gemm(stream, hk, 0, CF, W1s + (long)k * C1 * C1, 0, 256, nullptr,
                    nullptr, 0, 0, t, 0, 256, (int)MR, 256, 256, 1, 0);
        // 3b) u = relu(adj @ t + b1s[k])          (batched)
        launch_gemm(stream, adj, sAdj, 360, t, sT, 256, b1s + (long)k * C1,
                    nullptr, 0, 0, u, sT, 256, NODES, 256, 360, BATCH, 1);
        // 3c) t = u @ W2s[k]
        launch_gemm(stream, u, 0, 256, W2s + (long)k * C1 * C1, 0, 256, nullptr,
                    nullptr, 0, 0, t, 0, 256, (int)MR, 256, 256, 1, 0);
        // 3d) feats[k+1] = relu(adj @ t + b2s[k]) + h_k   (batched, residual)
        launch_gemm(stream, adj, sAdj, 360, t, sT, 256, b2s + (long)k * C1,
                    hk, sF, CF, feats + (long)(k + 1) * C1, sF, CF,
                    NODES, 256, 360, BATCH, 3);
    }

    // 4) fusion = relu(feats @ Wf + bf)          [23040,3584]x[3584,256]
    launch_gemm(stream, feats, 0, CF, Wf, 0, 256, bf,
                nullptr, 0, 0, fusion, 0, 256, (int)MR, 256, CF, 1, 1);
    // 5) hid = relu(fusion @ Wp1 + bp1)          [23040,256]x[256,64]
    launch_gemm(stream, fusion, 0, 256, Wp1, 0, 64, bp1,
                nullptr, 0, 0, hid, 0, 64, (int)MR, 64, 256, 1, 1);
    // 6) out = hid @ Wp2 + bp2                   [23040,64]x[64,2]
    final_linear<<<dim3((int)((MR + 255) / 256)), dim3(256), 0, stream>>>(
        hid, Wp2, bp2, out, (int)MR);
}
